// Attention_Dec_18262200942771
// MI455X (gfx1250) — matrix operand order repaired and checked by reading, not compiled, not run on hardware
//
#include <hip/hip_runtime.h>
#include <hip/hip_bf16.h>

// Attention_Dec: B=2, N=2048, DIM=1024, H=16, D=64
#define BB   2
#define NN   2048
#define DIMM 1024
#define HH   16
#define DD   64
#define SCALE 0.03125f   // DIM^-0.5

typedef _Float16 f16;
typedef __attribute__((ext_vector_type(16))) _Float16 v16h;
typedef __attribute__((ext_vector_type(8)))  _Float16 v8h;
typedef __attribute__((ext_vector_type(8)))  float    v8f;
typedef __attribute__((ext_vector_type(4)))  int      v4i;

#if defined(__has_builtin)
#if __has_builtin(__builtin_amdgcn_global_load_async_to_lds_b128)
#define USE_ASYNC 1
#endif
#endif

typedef __attribute__((address_space(1))) v4i as1_v4i;   // global (printed as __device__)
typedef __attribute__((address_space(3))) v4i as3_v4i;   // LDS    (printed as __shared__)

union V16H { v16h v; v8h h2[2]; f16 e[16]; };

// B-fragment (16-bit, KxN=32x16): per lane 16 contiguous K values of one column.
__device__ inline v16h loadB16_f16(const f16* p) {
  V16H u;
  u.h2[0] = *(const v8h*)(p);
  u.h2[1] = *(const v8h*)(p + 8);
  return u.v;
}

// A-fragment (16-bit, MxK=16x32): per lane 16 contiguous K values at 2*c0,
// c0 = (lane>>4)*8 (matches B-fragment k order).
__device__ inline v16h loadA_f16(const f16* rowp, int c0) {
  V16H u;
  u.h2[0] = *(const v8h*)(rowp + 2 * c0);
  u.h2[1] = *(const v8h*)(rowp + 2 * c0 + 8);
  return u.v;
}

#ifdef USE_ASYNC
// one 16-byte chunk per lane, global -> LDS via the async engine (ASYNCcnt)
__device__ inline void async_cp16(const f16* g, f16* l) {
  __builtin_amdgcn_global_load_async_to_lds_b128((as1_v4i*)g, (as3_v4i*)l, 0, 0);
}
#endif

// ---------------------------------------------------------------------------
// Kernel 0: elementwise f32 -> f16 conversion (activations + weights), so the
// GEMM hot loops are pure b128-load + wmma and re-read traffic is halved.
// ---------------------------------------------------------------------------
__global__ __launch_bounds__(256)
void cvt_f16_kernel(const float* __restrict__ in, f16* __restrict__ out, int n) {
  int i = (blockIdx.x * 256 + threadIdx.x) * 8;
  if (i + 8 <= n) {
#pragma unroll
    for (int t = 0; t < 8; ++t) out[i + t] = (f16)in[i + t];
  }
}

// ---------------------------------------------------------------------------
// Kernel 1: fused QKV projection (f16 in, f32 accumulate). Columns o in
// [0,3072): [0,1024)=Q, [1024,2048)=V (stored transposed [B,H,D,N]),
// [2048,3072)=K (tar @ W_k.T, stored [B,H,N,D]).
// ---------------------------------------------------------------------------
__global__ __launch_bounds__(256)
void qkv_proj_kernel(const f16* __restrict__ X16, const f16* __restrict__ T16,
                     const f16* __restrict__ Wqv16, const f16* __restrict__ Wk16,
                     f16* __restrict__ Qh, f16* __restrict__ Kh, f16* __restrict__ Vt) {
  const int lane   = threadIdx.x & 31;
  const int wave   = threadIdx.x >> 5;
  const int waveId = blockIdx.x * 8 + wave;
  const int mt = waveId % 256;        // 16-token row tile
  const int ct = waveId / 256;        // 64-wide column tile, [0,48)
  if (ct >= 48) return;

  const int half = lane >> 4;
  const int lm   = lane & 15;
  const int c0   = half * 8;

  const int seg = ct >> 4;            // 0=Q, 1=V, 2=K
  const f16* src = (seg == 2) ? T16 : X16;
  const f16* W   = (seg == 2) ? Wk16 : Wqv16;
  const int wrow0 = (seg == 2) ? (ct - 32) * 64 : ct * 64;

  const f16* arow = src + (size_t)(mt * 16 + lm) * DIMM;

  v8f zero = {};
  v8f acc[4];
#pragma unroll
  for (int s = 0; s < 4; ++s) acc[s] = zero;

  for (int k = 0; k < DIMM; k += 32) {
    __builtin_prefetch(arow + k + 64, 0, 3);      // global_prefetch_b8
    v16h a = loadA_f16(arow + k, c0);
#pragma unroll
    for (int s = 0; s < 4; ++s) {
      const f16* wr = W + (size_t)(wrow0 + s * 16 + lm) * DIMM + k + half * 16;
      v16h bf = loadB16_f16(wr);
      acc[s] = __builtin_amdgcn_wmma_f32_16x16x32_f16(false, a, false, bf,
                                                      (short)0, acc[s], false, false);
    }
  }

  // C layout: lane holds column (lm of subtile), rows r + 8*half in vgpr r.
#pragma unroll
  for (int s = 0; s < 4; ++s) {
#pragma unroll
    for (int r = 0; r < 8; ++r) {
      int t = mt * 16 + r + 8 * half;
      int b = t / NN, n = t % NN;
      int o = ct * 64 + s * 16 + lm;
      int c = o & 1023;
      int h = c >> 6, d = c & 63;
      f16 val = (f16)acc[s][r];
      if (seg == 0)      Qh[(((size_t)(b * HH + h)) * NN + n) * DD + d] = val;
      else if (seg == 1) Vt[(((size_t)(b * HH + h)) * DD + d) * NN + n] = val;
      else               Kh[(((size_t)(b * HH + h)) * NN + n) * DD + d] = val;
    }
  }
}

// ---------------------------------------------------------------------------
// Kernel 2: attention, online softmax, one wave per (b,h,16 tar rows).
// Q/V tiles staged to LDS with async global->LDS (ASYNCcnt), fragments pulled
// LDS->VGPR, next tile's fills overlapped with the 8 WMMAs + softmax VALU.
// ---------------------------------------------------------------------------
__global__ __launch_bounds__(128)
void attn_kernel(const f16* __restrict__ Qh, const f16* __restrict__ Kh,
                 const f16* __restrict__ Vt, const unsigned char* __restrict__ mask,
                 f16* __restrict__ Ows) {
  // per wave: Q tile 32x64 (4KB) + V tile 64x32 (4KB) + P tile 16x32 (1KB)
  __shared__ __align__(16) f16 smem[4][4608];
  const int lane   = threadIdx.x & 31;
  const int wave   = threadIdx.x >> 5;
  const int waveId = blockIdx.x * 4 + wave;
  const int it = waveId & 127;        // i tile (16 tar rows)
  const int bh = waveId >> 7;         // b*H + h in [0,32)
  const int b  = bh >> 4;
  const int h  = bh & 15;

  const int half = lane >> 4;
  const int lm   = lane & 15;
  const int c0   = half * 8;

  const f16* Kb = Kh + (size_t)bh * NN * DD;
  const f16* Qb = Qh + (size_t)bh * NN * DD;
  const f16* Vb = Vt + (size_t)bh * DD * NN;

  f16* Ql = smem[wave];
  f16* Vl = smem[wave] + 2048;
  f16* P  = smem[wave] + 4096;

  // K tile as A-frags (two 32-wide d chunks), reused across all j blocks
  const f16* krow = Kb + (size_t)(it * 16 + lm) * DD;
  v16h aK0 = loadA_f16(krow, c0);
  v16h aK1 = loadA_f16(krow + 32, c0);

  float mrow[8];
#pragma unroll
  for (int r = 0; r < 8; ++r) {
    int i = it * 16 + r + 8 * half;
    mrow[r] = (i == 0) ? 1.0f : (mask[b * (NN - 1) + i - 1] ? 1.0f : 0.0f);
  }

  v8f zero = {};
  v8f O[4];
#pragma unroll
  for (int s = 0; s < 4; ++s) O[s] = zero;
  float rmax[8], rsum[8];
#pragma unroll
  for (int r = 0; r < 8; ++r) { rmax[r] = -3.0e38f; rsum[r] = 0.0f; }

#ifdef USE_ASYNC
  // prefill tile 0: Q tile is one contiguous 4KB block; V tile is 64 rows x 64B
  {
    const f16* Qg = Qb;           // j0 = 0
    const f16* Vg = Vb;
#pragma unroll
    for (int t = 0; t < 8; ++t) {
      int idx = t * 32 + lane;
      async_cp16(Qg + idx * 8, Ql + idx * 8);
    }
#pragma unroll
    for (int t = 0; t < 8; ++t) {
      int idx = t * 32 + lane;
      int d = idx >> 2, sub = idx & 3;
      async_cp16(Vg + (size_t)d * NN + sub * 8, Vl + d * 32 + sub * 8);
    }
  }
#endif

  for (int j0 = 0; j0 < NN; j0 += 32) {
    v16h b00, b10, b01, b11, bv0, bv1, bv2, bv3;
#ifdef USE_ASYNC
    asm volatile("s_wait_asynccnt 0x0" ::: "memory");   // tile j0 resident in LDS
    // pull all fragments LDS -> VGPR (ds_load_b128)
    b00 = loadB16_f16(Ql + lm * 64 + half * 16);
    b10 = loadB16_f16(Ql + (16 + lm) * 64 + half * 16);
    b01 = loadB16_f16(Ql + lm * 64 + 32 + half * 16);
    b11 = loadB16_f16(Ql + (16 + lm) * 64 + 32 + half * 16);
    bv0 = loadB16_f16(Vl + (0 * 16 + lm) * 32 + half * 16);
    bv1 = loadB16_f16(Vl + (1 * 16 + lm) * 32 + half * 16);
    bv2 = loadB16_f16(Vl + (2 * 16 + lm) * 32 + half * 16);
    bv3 = loadB16_f16(Vl + (3 * 16 + lm) * 32 + half * 16);
    asm volatile("s_wait_dscnt 0x0" ::: "memory");      // LDS reads done: safe to refill
    if (j0 + 32 < NN) {                                 // overlap next fills with compute
      const f16* Qg = Qb + (size_t)(j0 + 32) * DD;
      const f16* Vg = Vb + (j0 + 32);
#pragma unroll
      for (int t = 0; t < 8; ++t) {
        int idx = t * 32 + lane;
        async_cp16(Qg + idx * 8, Ql + idx * 8);
      }
#pragma unroll
      for (int t = 0; t < 8; ++t) {
        int idx = t * 32 + lane;
        int d = idx >> 2, sub = idx & 3;
        async_cp16(Vg + (size_t)d * NN + sub * 8, Vl + d * 32 + sub * 8);
      }
    }
#else
    {
      const f16* q0 = Qb + (size_t)(j0 + lm) * DD + half * 16;
      const f16* q1 = q0 + 16 * DD;
      b00 = loadB16_f16(q0);
      b10 = loadB16_f16(q1);
      b01 = loadB16_f16(q0 + 32);
      b11 = loadB16_f16(q1 + 32);
      bv0 = loadB16_f16(Vb + (size_t)(0 * 16 + lm) * NN + j0 + half * 16);
      bv1 = loadB16_f16(Vb + (size_t)(1 * 16 + lm) * NN + j0 + half * 16);
      bv2 = loadB16_f16(Vb + (size_t)(2 * 16 + lm) * NN + j0 + half * 16);
      bv3 = loadB16_f16(Vb + (size_t)(3 * 16 + lm) * NN + j0 + half * 16);
    }
#endif

    // S = K_i (16xD) @ Q^T  -> 16x32 in two f32 accumulators
    v8f s0 = zero, s1 = zero;
    s0 = __builtin_amdgcn_wmma_f32_16x16x32_f16(false, aK0, false, b00, (short)0, s0, false, false);
    s1 = __builtin_amdgcn_wmma_f32_16x16x32_f16(false, aK0, false, b10, (short)0, s1, false, false);
    s0 = __builtin_amdgcn_wmma_f32_16x16x32_f16(false, aK1, false, b01, (short)0, s0, false, false);
    s1 = __builtin_amdgcn_wmma_f32_16x16x32_f16(false, aK1, false, b11, (short)0, s1, false, false);

    const int jc0 = j0 + lm;
    const float mj0 = (jc0 == 0) ? 1.0f : (mask[b * (NN - 1) + jc0 - 1] ? 1.0f : 0.0f);
    const float mj1 = mask[b * (NN - 1) + jc0 + 15] ? 1.0f : 0.0f;  // col jc0+16

    float alpha[8];
#pragma unroll
    for (int r = 0; r < 8; ++r) {
      float v0 = (mj0 * mrow[r] != 0.0f) ? s0[r] * SCALE : -1.0e30f;
      float v1 = (mj1 * mrow[r] != 0.0f) ? s1[r] * SCALE : -1.0e30f;
      float bm = fmaxf(v0, v1);
      bm = fmaxf(bm, __shfl_xor(bm, 1, 32));
      bm = fmaxf(bm, __shfl_xor(bm, 2, 32));
      bm = fmaxf(bm, __shfl_xor(bm, 4, 32));
      bm = fmaxf(bm, __shfl_xor(bm, 8, 32));
      float nm = fmaxf(rmax[r], bm);
      alpha[r] = __expf(rmax[r] - nm);
      rmax[r]  = nm;
      float p0 = __expf(v0 - nm);
      float p1 = __expf(v1 - nm);
      float ps = p0 + p1;
      ps += __shfl_xor(ps, 1, 32);
      ps += __shfl_xor(ps, 2, 32);
      ps += __shfl_xor(ps, 4, 32);
      ps += __shfl_xor(ps, 8, 32);
      rsum[r] = rsum[r] * alpha[r] + ps;
      int prow = r + 8 * half;
      P[prow * 32 + lm]      = (f16)p0;
      P[prow * 32 + lm + 16] = (f16)p1;
    }

#pragma unroll
    for (int s = 0; s < 4; ++s)
#pragma unroll
      for (int r = 0; r < 8; ++r) O[s][r] *= alpha[r];

    // O += P (16x32) @ V (32x64); A-frag re-read from LDS
    v16h aP = loadA_f16(P + lm * 32, c0);
    O[0] = __builtin_amdgcn_wmma_f32_16x16x32_f16(false, aP, false, bv0, (short)0, O[0], false, false);
    O[1] = __builtin_amdgcn_wmma_f32_16x16x32_f16(false, aP, false, bv1, (short)0, O[1], false, false);
    O[2] = __builtin_amdgcn_wmma_f32_16x16x32_f16(false, aP, false, bv2, (short)0, O[2], false, false);
    O[3] = __builtin_amdgcn_wmma_f32_16x16x32_f16(false, aP, false, bv3, (short)0, O[3], false, false);
  }

  // normalize and store token-major [B*N, DIM] f16
#pragma unroll
  for (int s = 0; s < 4; ++s)
#pragma unroll
    for (int r = 0; r < 8; ++r) {
      int i   = it * 16 + r + 8 * half;
      int col = h * 64 + s * 16 + lm;
      Ows[((size_t)(b * NN + i)) * DIMM + col] = (f16)(O[s][r] / rsum[r]);
    }
}

// ---------------------------------------------------------------------------
// Kernel 3: out = Ows @ W_out.T + b_out (fp32 output)
// ---------------------------------------------------------------------------
__global__ __launch_bounds__(256)
void out_proj_kernel(const f16* __restrict__ Ows, const f16* __restrict__ Wout16,
                     const float* __restrict__ b_out, float* __restrict__ out) {
  const int lane   = threadIdx.x & 31;
  const int wave   = threadIdx.x >> 5;
  const int waveId = blockIdx.x * 8 + wave;
  const int mt = waveId & 255;
  const int ct = waveId >> 8;          // [0,16)
  const int half = lane >> 4;
  const int lm   = lane & 15;
  const int c0   = half * 8;

  const f16* arow = Ows + (size_t)(mt * 16 + lm) * DIMM;

  v8f zero = {};
  v8f acc[4];
#pragma unroll
  for (int s = 0; s < 4; ++s) acc[s] = zero;

  for (int k = 0; k < DIMM; k += 32) {
    __builtin_prefetch(arow + k + 64, 0, 3);
    v16h a = loadA_f16(arow + k, c0);
#pragma unroll
    for (int s = 0; s < 4; ++s) {
      const f16* wr = Wout16 + (size_t)(ct * 64 + s * 16 + lm) * DIMM + k + half * 16;
      v16h bf = loadB16_f16(wr);
      acc[s] = __builtin_amdgcn_wmma_f32_16x16x32_f16(false, a, false, bf,
                                                      (short)0, acc[s], false, false);
    }
  }

#pragma unroll
  for (int s = 0; s < 4; ++s) {
    int o = ct * 64 + s * 16 + lm;
    float bias = b_out[o];
#pragma unroll
    for (int r = 0; r < 8; ++r) {
      int t = mt * 16 + r + 8 * half;
      out[(size_t)t * DIMM + o] = acc[s][r] + bias;
    }
  }
}

// ---------------------------------------------------------------------------
extern "C" void kernel_launch(void* const* d_in, const int* in_sizes, int n_in,
                              void* d_out, int out_size, void* d_ws, size_t ws_size,
                              hipStream_t stream) {
  (void)in_sizes; (void)n_in; (void)out_size; (void)ws_size;
  const float* x     = (const float*)d_in[0];
  const float* tar   = (const float*)d_in[1];
  const unsigned char* mask = (const unsigned char*)d_in[2];
  const float* W_qv  = (const float*)d_in[3];
  const float* W_k   = (const float*)d_in[4];
  const float* W_out = (const float*)d_in[5];
  const float* b_out = (const float*)d_in[6];
  float* out = (float*)d_out;

  const size_t M1 = (size_t)1 << 20;
  f16* Qh     = (f16*)d_ws;            // 4M f16
  f16* Kh     = Qh + 4 * M1;           // 4M
  f16* Vt     = Kh + 4 * M1;           // 4M   [B,H,D,N]
  f16* Ows    = Vt + 4 * M1;           // 4M   [B*N, DIM]
  f16* X16    = Ows + 4 * M1;          // 4M
  f16* T16    = X16 + 4 * M1;          // 4M
  f16* Wqv16  = T16 + 4 * M1;          // 2M
  f16* Wk16   = Wqv16 + 2 * M1;        // 1M
  f16* Wout16 = Wk16 + 1 * M1;         // 1M  (total 28M f16 = 56MB)

  // f32 -> f16 conversions (each n is a multiple of 2048)
  cvt_f16_kernel<<<(int)(4 * M1 / 2048), 256, 0, stream>>>(x,     X16,    (int)(4 * M1));
  cvt_f16_kernel<<<(int)(4 * M1 / 2048), 256, 0, stream>>>(tar,   T16,    (int)(4 * M1));
  cvt_f16_kernel<<<(int)(2 * M1 / 2048), 256, 0, stream>>>(W_qv,  Wqv16,  (int)(2 * M1));
  cvt_f16_kernel<<<(int)(1 * M1 / 2048), 256, 0, stream>>>(W_k,   Wk16,   (int)(1 * M1));
  cvt_f16_kernel<<<(int)(1 * M1 / 2048), 256, 0, stream>>>(W_out, Wout16, (int)(1 * M1));

  // 256 row-tiles x 48 col-tiles = 12288 waves -> 1536 blocks of 8 waves
  qkv_proj_kernel<<<1536, 256, 0, stream>>>(X16, T16, Wqv16, Wk16, Qh, Kh, Vt);
  // 2*16*128 = 4096 waves -> 1024 blocks of 4 waves (9KB LDS per wave)
  attn_kernel<<<1024, 128, 0, stream>>>(Qh, Kh, Vt, mask, Ows);
  // 256 x 16 = 4096 waves -> 512 blocks
  out_proj_kernel<<<512, 256, 0, stream>>>(Ows, Wout16, b_out, out);
}
